// FactorizedLSTMCell_19009525252388
// MI455X (gfx1250) — matrix operand order repaired and checked by reading, not compiled, not run on hardware
//
#include <hip/hip_runtime.h>
#include <cstdint>
#include <cstddef>

// ---------------------------------------------------------------------------
// FactorizedLSTMCell for MI455X (gfx1250), bf16 WMMA pipeline.
//   stage 1: T_g[8192,512]  = combined[8192,4096] @ U_g[4096,512]   (x4 gates)
//   stage 2: G_g[8192,2048] = T_g @ V_g + b_g  -> fused activations -> h_new,c_new
// GEMMs via v_wmma_f32_16x16x32_bf16 (wave32).
// Stage-1 tile staging: TDM tensor_load_to_lds (TENSORcnt), double-buffered,
//   with hardware LDS padding producing the bank-friendly tile pitch.
// Stage-2 tile staging: global_load_async_to_lds_b128 (ASYNCcnt).
// ---------------------------------------------------------------------------

typedef __bf16 bf16_t;
typedef __attribute__((ext_vector_type(16))) __bf16 v16bf;
typedef __attribute__((ext_vector_type(8)))  float  v8f;
typedef __attribute__((ext_vector_type(4)))  unsigned v4u;
typedef __attribute__((ext_vector_type(8)))  unsigned v8u;

#define B_DIM   8192
#define IN_DIM  2048
#define HID_DIM 2048
#define K1      4096   // IN + HID
#define RANK    512

#define PA 40   // LDS pitch for 128x32 A tiles (bf16): 64B row + 16B pad
#define PB 72   // LDS pitch for 32x64  B tiles (bf16): 128B row + 16B pad

// float -> bf16, round-to-nearest-even, pure bit ops
__device__ __forceinline__ bf16_t f2bf(float f) {
  unsigned u = __builtin_bit_cast(unsigned, f);
  u += 0x7FFFu + ((u >> 16) & 1u);
  return __builtin_bit_cast(bf16_t, (unsigned short)(u >> 16));
}

__device__ __forceinline__ float sigmoidf_(float x) {
  return 1.0f / (1.0f + __expf(-x));
}
__device__ __forceinline__ float tanhf_(float x) {
  float e = __expf(-2.0f * x);
  return (1.0f - e) / (1.0f + e);
}

// --- CDNA5 async global->LDS copy (ASYNCcnt), ISA 10 / CDNA5 ISA ---
__device__ __forceinline__ void async_copy_b128(void* lds_ptr, const void* gptr) {
  unsigned lds_off = (unsigned)(uintptr_t)lds_ptr;     // low 32 bits = LDS offset
  unsigned long long ga = (unsigned long long)(uintptr_t)gptr;
  asm volatile("global_load_async_to_lds_b128 %0, %1, off"
               :: "v"(lds_off), "v"(ga)
               : "memory");
}
__device__ __forceinline__ void wait_async0() {
  asm volatile("s_wait_asynccnt 0x0" ::: "memory");
}

// --- CDNA5 Tensor Data Mover: 2D tile load, D# per ISA 08 §8.3/8.4 ----------
// Group0: [1:0]=count=1, [63:32]=lds_addr, [120:64]=global_addr, [127:126]=type 2
// Group1: data_size=2B, pad_enable + pad_interval/amount (HW LDS pitch pad),
//         tensor_dim = tile_dim (tile-local walk, never OOB), dim0 stride = row pitch.
// pad_interval_code: interval = (1<<code) 8-byte units; pad_amount_code: (code+1) DWORDs.
__device__ __forceinline__ void tdm_load_tile_2d(
    void* lds_base, const void* gaddr,
    unsigned tile_w, unsigned tile_h,           // in elements (2B each)
    unsigned long long row_stride_elems,
    unsigned pad_interval_code, unsigned pad_amount_code)
{
  unsigned lds = (unsigned)(uintptr_t)lds_base;
  unsigned long long ga = (unsigned long long)(uintptr_t)gaddr;
  v4u g0;
  g0[0] = 1u;                                          // count=1 (valid user D#)
  g0[1] = lds;                                         // lds_addr (bytes)
  g0[2] = (unsigned)(ga & 0xFFFFFFFFull);              // global_addr[31:0]
  g0[3] = (unsigned)((ga >> 32) & 0x01FFFFFFull)       // global_addr[56:32]
        | 0x80000000u;                                 // type = 2 ("image")
  v8u g1;
  g1[0] = (1u << 16)                                   // data_size = 1 -> 2 bytes
        | (1u << 20)                                   // pad_enable
        | (pad_interval_code << 22)
        | (pad_amount_code << 25);
  g1[1] = (tile_w & 0xFFFFu) << 16;                    // abar=0 | tensor_dim0 lo16
  g1[2] = (tile_w >> 16) | ((tile_h & 0xFFFFu) << 16); // tensor_dim0 hi | dim1 lo
  g1[3] = (tile_h >> 16) | (tile_w << 16);             // tensor_dim1 hi | tile_dim0
  g1[4] = tile_h;                                      // tile_dim1 | tile_dim2=0
  g1[5] = (unsigned)(row_stride_elems & 0xFFFFFFFFull);// tensor_dim0_stride lo32
  g1[6] = (unsigned)((row_stride_elems >> 32) & 0xFFFFull); // stride hi16 | d1s=0
  g1[7] = 0u;
  asm volatile("tensor_load_to_lds %0, %1" :: "s"(g0), "s"(g1) : "memory");
}
__device__ __forceinline__ void wait_tensor0() {
  __builtin_amdgcn_s_wait_tensorcnt(0);
}

union FragBF { v16bf v; uint4 q[2]; };

// A fragment (16x32, MxK, 16-bit): lane L holds row m=L&15; half h=L>>4.
// element e (0..15): K = (e&7) + 8*h + 16*(e>>3)   [ISA 7.12.2]
__device__ __forceinline__ v16bf load_a_frag(const bf16_t* tile, int pitch, int m0, int lane) {
  const int m  = m0 + (lane & 15);
  const int kh = (lane >> 4) << 3;            // 0 or 8
  const bf16_t* p = tile + m * pitch;
  FragBF f;
  f.q[0] = *(const uint4*)(p + kh);           // K = kh .. kh+7       -> e0..7
  f.q[1] = *(const uint4*)(p + 16 + kh);      // K = 16+kh .. 16+kh+7 -> e8..15
  return f.v;
}

// B fragment (32x16, KxN, 16-bit): lane L holds column n=L&15; half h=L>>4.
// element e (0..15): K = (e&7) + 8*h + 16*(e>>3)
__device__ __forceinline__ v16bf load_b_frag(const bf16_t* tile, int pitch, int n0, int lane) {
  const int kh = (lane >> 4) << 3;
  const bf16_t* p = tile + n0 + (lane & 15);
  FragBF f;
#pragma unroll
  for (int e = 0; e < 8; ++e) {
    f.v[e]     = p[(kh + e) * pitch];
    f.v[e + 8] = p[(16 + kh + e) * pitch];
  }
  return f.v;
}

// ---------------------------------------------------------------------------
// Conversion kernels
// ---------------------------------------------------------------------------
__global__ __launch_bounds__(256) void pack_combined_kernel(
    const float* __restrict__ x, const float* __restrict__ h,
    bf16_t* __restrict__ dst) {
  size_t i = ((size_t)blockIdx.x * 256 + threadIdx.x) * 8;   // elem idx in [B,4096]
  if (i >= (size_t)B_DIM * K1) return;
  int row = (int)(i >> 12);
  int col = (int)(i & 4095);
  const float* src = (col < IN_DIM)
      ? (x + (size_t)row * IN_DIM + col)
      : (h + (size_t)row * HID_DIM + (col - IN_DIM));
  float4 f0 = *(const float4*)(src);
  float4 f1 = *(const float4*)(src + 4);
  union { bf16_t b[8]; uint4 q; } pk;
  pk.b[0] = f2bf(f0.x); pk.b[1] = f2bf(f0.y); pk.b[2] = f2bf(f0.z); pk.b[3] = f2bf(f0.w);
  pk.b[4] = f2bf(f1.x); pk.b[5] = f2bf(f1.y); pk.b[6] = f2bf(f1.z); pk.b[7] = f2bf(f1.w);
  *(uint4*)(dst + i) = pk.q;
}

__global__ __launch_bounds__(256) void convert_f32_bf16_kernel(
    const float* __restrict__ src, bf16_t* __restrict__ dst, long n) {
  size_t i = ((size_t)blockIdx.x * 256 + threadIdx.x) * 8;
  if (i >= (size_t)n) return;
  float4 f0 = *(const float4*)(src + i);
  float4 f1 = *(const float4*)(src + i + 4);
  union { bf16_t b[8]; uint4 q; } pk;
  pk.b[0] = f2bf(f0.x); pk.b[1] = f2bf(f0.y); pk.b[2] = f2bf(f0.z); pk.b[3] = f2bf(f0.w);
  pk.b[4] = f2bf(f1.x); pk.b[5] = f2bf(f1.y); pk.b[6] = f2bf(f1.z); pk.b[7] = f2bf(f1.w);
  *(uint4*)(dst + i) = pk.q;
}

// ---------------------------------------------------------------------------
// Stage 1 GEMM: T_g = combined @ U_g   (M=8192, N=512, K=4096), gate = blockIdx.z
// Block tile 128(M) x 64(N), 256 threads = 8 waves; wave computes 64x16.
// TDM staging: wave 0 DMAs the A tile, wave 1 DMAs the B tile. Double-buffered.
// ---------------------------------------------------------------------------
__global__ __launch_bounds__(256) void gemm1_kernel(
    const bf16_t* __restrict__ A,     // combined [8192,4096]
    const bf16_t* __restrict__ Uall,  // 4 x [4096,512]
    bf16_t* __restrict__ Tall)        // 4 x [8192,512]
{
  __shared__ __align__(16) bf16_t As[2][128][PA];
  __shared__ __align__(16) bf16_t Bs[2][32][PB];

  const int g = blockIdx.z;
  const bf16_t* Ug = Uall + (size_t)g * K1 * RANK;
  bf16_t*       Tg = Tall + (size_t)g * B_DIM * RANK;

  const int nBase = blockIdx.x * 64;
  const int mBase = blockIdx.y * 128;
  const int tid   = threadIdx.x;
  const int lane  = tid & 31;
  const int wave  = tid >> 5;
  const int waveN = wave & 3;   // 4 x 16-col strips
  const int waveM = wave >> 2;  // 2 x 64-row strips

  v8f acc[4];
#pragma unroll
  for (int i = 0; i < 4; ++i) acc[i] = (v8f){0.f,0.f,0.f,0.f,0.f,0.f,0.f,0.f};

  auto issue_tile = [&](int buf, int ko) {
    if (wave == 0) {
      // A tile: 128 rows x 32 cols, row pitch K1; pad 16 DW interval, +4 DW (->PA)
      tdm_load_tile_2d(&As[buf][0][0],
                       A + (size_t)mBase * K1 + ko,
                       /*tile_w=*/32, /*tile_h=*/128, /*stride=*/K1,
                       /*pad_interval=*/3, /*pad_amount=*/3);
    } else if (wave == 1) {
      // B tile: 32 rows x 64 cols, row pitch RANK; pad 32 DW interval, +4 DW (->PB)
      tdm_load_tile_2d(&Bs[buf][0][0],
                       Ug + (size_t)ko * RANK + nBase,
                       /*tile_w=*/64, /*tile_h=*/32, /*stride=*/RANK,
                       /*pad_interval=*/4, /*pad_amount=*/3);
    }
  };

  issue_tile(0, 0);
  int cur = 0;
#pragma unroll 2
  for (int ko = 0; ko < K1; ko += 32) {
    wait_tensor0();       // issuing waves: DMA into buf[cur] landed (no-op others)
    __syncthreads();      // whole tile visible to all waves
    if (ko + 32 < K1) issue_tile(cur ^ 1, ko + 32);   // overlap next slab

    v16bf bfrag = load_b_frag(&Bs[cur][0][0], PB, waveN * 16, lane);
#pragma unroll
    for (int mi = 0; mi < 4; ++mi) {
      v16bf afrag = load_a_frag(&As[cur][0][0], PA, waveM * 64 + mi * 16, lane);
      acc[mi] = __builtin_amdgcn_wmma_f32_16x16x32_bf16(
          false, afrag, false, bfrag, (short)0, acc[mi], false, false);
    }
    __syncthreads();      // all reads of buf[cur] done before it is refilled
    cur ^= 1;
  }

  // Epilogue: D layout (16x16 f32): lane L -> n = L&15; VGPR r -> m = r + 8*(L>>4)
  const int nn   = nBase + waveN * 16 + (lane & 15);
  const int half = lane >> 4;
#pragma unroll
  for (int mi = 0; mi < 4; ++mi) {
    int mfr = mBase + waveM * 64 + mi * 16 + 8 * half;
#pragma unroll
    for (int r = 0; r < 8; ++r) {
      Tg[(size_t)(mfr + r) * RANK + nn] = f2bf(acc[mi][r]);
    }
  }
}

// ---------------------------------------------------------------------------
// Stage 2 fused: for all 4 gates G_g = T_g @ V_g + b_g, then LSTM elementwise.
// M=8192, N=2048, K=512. Block tile 128x64; wave computes 64x16 for 4 gates.
// Async global->LDS staging (ASYNCcnt), single buffer (58 KB LDS).
// ---------------------------------------------------------------------------
__global__ __launch_bounds__(256) void gemm2_fused_kernel(
    const bf16_t* __restrict__ Tall,  // 4 x [8192,512]
    const bf16_t* __restrict__ Vall,  // 4 x [512,2048]
    const float* __restrict__ cin,    // [8192,2048]
    const float* __restrict__ bi, const float* __restrict__ bff,
    const float* __restrict__ bc_, const float* __restrict__ bo,
    float* __restrict__ out)          // h_new then c_new, each [8192,2048]
{
  __shared__ __align__(16) bf16_t As[4][128][PA];  // 40 KB
  __shared__ __align__(16) bf16_t Bs[4][32][PB];   // 18 KB

  const int nBase = blockIdx.x * 64;
  const int mBase = blockIdx.y * 128;
  const int tid   = threadIdx.x;
  const int lane  = tid & 31;
  const int wave  = tid >> 5;
  const int waveN = wave & 3;
  const int waveM = wave >> 2;

  const int ar0 = (tid) >> 2,        ac0 = ((tid) & 3) << 3;
  const int ar1 = (tid + 256) >> 2,  ac1 = ((tid + 256) & 3) << 3;
  const int br  = tid >> 3,          bc  = (tid & 7) << 3;

  v8f acc[4][4];   // [gate][m-frag]
#pragma unroll
  for (int gq = 0; gq < 4; ++gq)
#pragma unroll
    for (int mi = 0; mi < 4; ++mi)
      acc[gq][mi] = (v8f){0.f,0.f,0.f,0.f,0.f,0.f,0.f,0.f};

#pragma unroll 4
  for (int ko = 0; ko < RANK; ko += 32) {
    __syncthreads();   // previous iteration's reads complete before refill
#pragma unroll
    for (int gq = 0; gq < 4; ++gq) {
      const bf16_t* Tg = Tall + (size_t)gq * B_DIM * RANK;
      const bf16_t* Vg = Vall + (size_t)gq * RANK * HID_DIM;
      async_copy_b128(&As[gq][ar0][ac0], Tg + (size_t)(mBase + ar0) * RANK + ko + ac0);
      async_copy_b128(&As[gq][ar1][ac1], Tg + (size_t)(mBase + ar1) * RANK + ko + ac1);
      async_copy_b128(&Bs[gq][br][bc],   Vg + (size_t)(ko + br) * HID_DIM + nBase + bc);
    }
    // prefetch next K-slab into cache (single-buffered here)
    if (ko + 32 < RANK) {
      __builtin_prefetch(Tall + (size_t)(mBase + ar0) * RANK + ko + 32, 0, 0);
      __builtin_prefetch(Vall + (size_t)(ko + 32 + br) * HID_DIM + nBase, 0, 0);
    }
    wait_async0();
    __syncthreads();

#pragma unroll
    for (int gq = 0; gq < 4; ++gq) {
      v16bf bfrag = load_b_frag(&Bs[gq][0][0], PB, waveN * 16, lane);
#pragma unroll
      for (int mi = 0; mi < 4; ++mi) {
        v16bf afrag = load_a_frag(&As[gq][0][0], PA, waveM * 64 + mi * 16, lane);
        acc[gq][mi] = __builtin_amdgcn_wmma_f32_16x16x32_bf16(
            false, afrag, false, bfrag, (short)0, acc[gq][mi], false, false);
      }
    }
  }

  // Fused LSTM epilogue. Gate order: 0=i, 1=f, 2=c(hat), 3=o.
  const int nn   = nBase + waveN * 16 + (lane & 15);
  const int half = lane >> 4;
  const float Bi = bi[nn], Bf = bff[nn], Bc = bc_[nn], Bo = bo[nn];
  float* outH = out;
  float* outC = out + (size_t)B_DIM * HID_DIM;

#pragma unroll
  for (int mi = 0; mi < 4; ++mi) {
    int mfr = mBase + waveM * 64 + mi * 16 + 8 * half;
#pragma unroll
    for (int r = 0; r < 8; ++r) {
      size_t idx = (size_t)(mfr + r) * HID_DIM + nn;
      float gi = sigmoidf_(acc[0][mi][r] + Bi);
      float gf = sigmoidf_(acc[1][mi][r] + Bf);
      float gc = tanhf_   (acc[2][mi][r] + Bc);
      float go = sigmoidf_(acc[3][mi][r] + Bo);
      float cold = cin[idx];
      float cnew = gf * cold + gi * gc;
      float hnew = go * tanhf_(cnew);
      outH[idx] = hnew;
      outC[idx] = cnew;
    }
  }
}

// ---------------------------------------------------------------------------
// Launch
// ---------------------------------------------------------------------------
extern "C" void kernel_launch(void* const* d_in, const int* in_sizes, int n_in,
                              void* d_out, int out_size, void* d_ws, size_t ws_size,
                              hipStream_t stream) {
  const float* x = (const float*)d_in[0];
  const float* h = (const float*)d_in[1];
  const float* c = (const float*)d_in[2];
  const float* U[4] = {(const float*)d_in[3], (const float*)d_in[6],
                       (const float*)d_in[9], (const float*)d_in[12]};
  const float* V[4] = {(const float*)d_in[4], (const float*)d_in[7],
                       (const float*)d_in[10], (const float*)d_in[13]};
  const float* bias[4] = {(const float*)d_in[5], (const float*)d_in[8],
                          (const float*)d_in[11], (const float*)d_in[14]};

  // Workspace layout (bf16): combined | 4xU | 4xV | 4xT   (~120 MiB total)
  bf16_t* combined = (bf16_t*)d_ws;
  bf16_t* Ubf = combined + (size_t)B_DIM * K1;
  bf16_t* Vbf = Ubf + (size_t)4 * K1 * RANK;
  bf16_t* Tbf = Vbf + (size_t)4 * RANK * HID_DIM;

  // 1) convert inputs to bf16
  {
    size_t n = (size_t)B_DIM * K1;                 // 33,554,432
    pack_combined_kernel<<<dim3((unsigned)(n / 8 / 256)), dim3(256), 0, stream>>>(
        x, h, combined);
  }
  for (int g = 0; g < 4; ++g) {
    long nU = (long)K1 * RANK;                     // 2,097,152
    convert_f32_bf16_kernel<<<dim3((unsigned)(nU / 8 / 256)), dim3(256), 0, stream>>>(
        U[g], Ubf + (size_t)g * K1 * RANK, nU);
    long nV = (long)RANK * HID_DIM;                // 1,048,576
    convert_f32_bf16_kernel<<<dim3((unsigned)(nV / 8 / 256)), dim3(256), 0, stream>>>(
        V[g], Vbf + (size_t)g * RANK * HID_DIM, nV);
  }

  // 2) stage-1 GEMMs (all 4 gates via blockIdx.z)
  gemm1_kernel<<<dim3(RANK / 64, B_DIM / 128, 4), dim3(256), 0, stream>>>(
      combined, Ubf, Tbf);

  // 3) stage-2 GEMM + fused LSTM elementwise
  gemm2_fused_kernel<<<dim3(HID_DIM / 64, B_DIM / 128), dim3(256), 0, stream>>>(
      Tbf, Vbf, c, bias[0], bias[1], bias[2], bias[3], (float*)d_out);
}
